// Correlation_62560493634034
// MI455X (gfx1250) — matrix operand order repaired and checked by reading, not compiled, not run on hardware
//
#include <hip/hip_runtime.h>

// FlowNet correlation, MAX_DISP=4, bf16 WMMA (v_wmma_f32_16x16x32_bf16) with
// block-wide LDS staging of the in2 row (packed bf16).
// B=8, C=256, H=96, W=128 -> out [8, 81, 96, 128] fp32.
//
// Per 16-pixel strip of input1 at (b,h) and per dy, the needed products are
// the +-4 diagonal band of a 16x24 GEMM between the in1 strip (16xC) and a
// 24-col in2 window (Cx24): two 16x16x32-bf16 WMMA tiles, K in 8 chunks.
// The in2 row h2 (256ch x 128col) is staged once per block per dy into LDS as
// bf16 packed pairs, layout [col][k/2] (pitch ROWP dwords), so each B-fragment
// is 16 consecutive bf16 -> two aligned 16B LDS loads.

#define BB   8
#define CC   256
#define HH   96
#define WW   128
#define DISP 4
#define ND   9          // 2*DISP+1
#define NK   8          // C / 32 k-chunks
#define ROWP 132        // LDS pitch in dwords per col (128 k-pairs + 4 pad)

typedef __attribute__((ext_vector_type(16))) __bf16 v16bf;
typedef __attribute__((ext_vector_type(8)))  __bf16 v8bf;
typedef __attribute__((ext_vector_type(4)))  __bf16 v4bf;
typedef __attribute__((ext_vector_type(8)))  float  v8f;
typedef __attribute__((ext_vector_type(4)))  float  v4f;

__global__ __launch_bounds__(256)
void corr_wmma_kernel(const float* __restrict__ in1,
                      const float* __restrict__ in2,
                      float* __restrict__ out) {
    // in2 row h2, bf16-packed: bf16 index = (col*ROWP + k/2)*2 + (k&1)
    __shared__ __align__(16) __bf16 s_in2[WW * ROWP * 2];
    __shared__ float band[8][ND][16];          // per-wave 9x16 band staging

    const int HW   = HH * WW;
    const int bh   = blockIdx.x;               // 0 .. B*H-1
    const int b    = bh / HH;
    const int h    = bh % HH;
    const int wave = threadIdx.x >> 5;         // 0..7 -> column tile
    const int lane = threadIdx.x & 31;
    const int n16  = lane & 15;                // A-row / B-col / C-col role
    const int hi   = lane >> 4;                // half-wave select
    const int w0   = wave * 16;

    // ---- A fragments (in1 strip, all K), one-time global gather ------------
    // A[m][k] = in1[b, k, h, w0+m]; lane m = n16.
    // Element e <-> K = kc*32 + e + 16*hi   (contiguous order, matches B).
    const float* a_base = in1 + (size_t)b * CC * HW + (size_t)h * WW + (w0 + n16);
    v16bf afrag[NK];
#pragma unroll
    for (int kc = 0; kc < NK; ++kc) {
        const int kb = kc * 32 + hi * 16;
#pragma unroll
        for (int e = 0; e < 16; ++e) {
            const int k = kb + e;
            afrag[kc][e] = (__bf16)a_base[(size_t)k * HW];
        }
    }

    // ---- B window columns (fixed per lane) ---------------------------------
    // tile0: cols w0-4 .. w0+11, tile1: cols w0+12 .. w0+27. OOB columns are
    // clamped (load real data) and zeroed at extraction via the fused scale.
    const int col0r = w0 - 4 + n16;
    const int col1r = w0 + 12 + n16;
    const bool v0 = (col0r >= 0) && (col0r < WW);
    const bool v1 = (col1r < WW);
    const int col0 = v0 ? col0r : 0;
    const int col1 = v1 ? col1r : (WW - 1);

    const float inv_c = 1.0f / (float)CC;
    const float s0 = v0 ? inv_c : 0.0f;        // fused normalize + col-zero
    const float s1 = v1 ? inv_c : 0.0f;

    const float* b2_base = in2 + (size_t)b * CC * HW;

#pragma unroll 1
    for (int dy = 0; dy < ND; ++dy) {
        const int h2 = h + dy - DISP;
        const bool rowok = (h2 >= 0) && (h2 < HH);   // uniform across block

        // ---- Cooperative stage: in2[b, :, h2, :] -> LDS bf16 ---------------
        if (rowok) {
            const float* rbase = b2_base + (size_t)h2 * WW;
            const int c0 = lane * 4;                   // 4 cols per lane
#pragma unroll
            for (int j = 0; j < 8; ++j) {
                const int kq = wave + 8 * j;           // k-quad 0..63
                const float* p = rbase + (size_t)(kq * 4) * HW + c0;
                const v4f f0 = *(const v4f*)(p);               // k = 4kq
                const v4f f1 = *(const v4f*)(p + HW);          // k = 4kq+1
                const v4f f2 = *(const v4f*)(p + 2 * HW);      // k = 4kq+2
                const v4f f3 = *(const v4f*)(p + 3 * HW);      // k = 4kq+3
#pragma unroll
                for (int c = 0; c < 4; ++c) {
                    v4bf pk;
                    pk[0] = (__bf16)f0[c];
                    pk[1] = (__bf16)f1[c];
                    pk[2] = (__bf16)f2[c];
                    pk[3] = (__bf16)f3[c];
                    // dword addr = col*ROWP + 2*kq  (even -> 8B aligned)
                    *(v4bf*)&s_in2[((size_t)(c0 + c) * ROWP + 2 * kq) * 2] = pk;
                }
            }
        }
        __syncthreads();   // staging visible (uniform branch above)

        v8f acc0 = {};
        v8f acc1 = {};
        if (rowok) {
#pragma unroll
            for (int kc = 0; kc < NK; ++kc) {
                // B element e <-> K = kc*32 + e + 16*hi: 16 consecutive bf16
                // at fixed col -> two aligned 16B LDS loads.
                const int kb2 = kc * 16 + hi * 8;      // K/2 base, mult of 4
                const __bf16* p0 = &s_in2[((size_t)col0 * ROWP + kb2) * 2];
                const __bf16* p1 = &s_in2[((size_t)col1 * ROWP + kb2) * 2];
                const v8bf a0 = *(const v8bf*)p0;
                const v8bf a1 = *(const v8bf*)(p0 + 8);
                const v8bf b0 = *(const v8bf*)p1;
                const v8bf b1 = *(const v8bf*)(p1 + 8);
                v16bf bf0, bf1;
#pragma unroll
                for (int e = 0; e < 8; ++e) {
                    bf0[e] = a0[e]; bf0[e + 8] = a1[e];
                    bf1[e] = b0[e]; bf1[e + 8] = b1[e];
                }
                acc0 = __builtin_amdgcn_wmma_f32_16x16x32_bf16(
                           false, afrag[kc], false, bf0, (short)0, acc0, false, false);
                acc1 = __builtin_amdgcn_wmma_f32_16x16x32_bf16(
                           false, afrag[kc], false, bf1, (short)0, acc1, false, false);
            }
        }
        // !rowok: accs stay zero -> zero-padded rows, correct.

        // ---- Extract +-4 diagonal band ------------------------------------
        // C/D layout: lane = column N=n16; VGPR r = row M = r + 8*hi.
        // tile0: di = n16 - m;  tile1: di = n16 + 16 - m.  s0/s1 zero OOB dx.
#pragma unroll
        for (int r = 0; r < 8; ++r) {
            const int m = r + hi * 8;
            const int di0 = n16 - m;
            if (di0 >= 0 && di0 <= 8) band[wave][di0][m] = acc0[r] * s0;
            const int di1 = n16 + 16 - m;
            if (di1 >= 0 && di1 <= 8) band[wave][di1][m] = acc1[r] * s1;
        }
        // Same-wave LDS write->read is in-order; no block barrier needed here.

        // ---- Store 9 coalesced rows of 16 floats ---------------------------
        for (int i = lane; i < ND * 16; i += 32) {
            const int di = i >> 4;
            const int m  = i & 15;
            const size_t oidx =
                (((size_t)b * (ND * ND) + (size_t)(dy * ND + di)) * HH + h) * WW
                + (w0 + m);
            out[oidx] = band[wave][di][m];
        }
        __syncthreads();   // all fragment reads done before next dy restages
    }
}

extern "C" void kernel_launch(void* const* d_in, const int* in_sizes, int n_in,
                              void* d_out, int out_size, void* d_ws, size_t ws_size,
                              hipStream_t stream) {
    const float* in1 = (const float*)d_in[0];
    const float* in2 = (const float*)d_in[1];
    float* out = (float*)d_out;
    (void)in_sizes; (void)n_in; (void)out_size; (void)d_ws; (void)ws_size;

    dim3 grid(BB * HH);   // 768 blocks; 8 waves/block = one 16-px tile per wave
    dim3 block(256);
    corr_wmma_kernel<<<grid, block, 0, stream>>>(in1, in2, out);
}
